// FLASH_ShareA_FFConvM_9028021256374
// MI455X (gfx1250) — compile-verified, matrix operand order checked by reading, not run on hardware
//
#include <hip/hip_runtime.h>

typedef unsigned short u16;
typedef __attribute__((ext_vector_type(8)))  float  v8f;
typedef __attribute__((ext_vector_type(16))) __bf16 v16bf;
typedef __attribute__((ext_vector_type(8)))  __bf16 v8bf;

// ---------------------------------------------------------------- helpers

__device__ __forceinline__ u16 f2bf(float f) {
  union { float f; unsigned u; } c; c.f = f;
  unsigned u = c.u + 0x7FFFu + ((c.u >> 16) & 1u);  // round-to-nearest-even
  return (u16)(u >> 16);
}

__device__ __forceinline__ v16bf make_frag(v8bf lo, v8bf hi) {
  v16bf f;
#pragma unroll
  for (int i = 0; i < 8; ++i) { f[i] = lo[i]; f[i + 8] = hi[i]; }
  return f;
}

// ---------------------------------------------------------------- WMMA GEMM
// C[z] = epilogue(A[z] @ Bt[z]^T)
//   A  : bf16 [M,K] row-major (lda)
//   Bt : bf16 [N,K] row-major (ldb)  (i.e. B pre-transposed, K contiguous)
// modes: 0: C=acc*alpha (f32)        1: C=silu(acc+bias[n]) (f32)
//        2: C=bf16(relu(acc*alpha)^2, zero if col>row)   3: C+=acc (f32)
// B tiles are double-buffered in LDS via async DMA (GLOBAL_LOAD_ASYNC_TO_LDS,
// ASYNCcnt), overlapping HBM traffic for tile k+1 with WMMAs on tile k.
__global__ void __launch_bounds__(256)
wmma_gemm_kernel(const u16* __restrict__ A, long long sAz, int lda,
                 const u16* __restrict__ Bt, long long sBz, int ldb,
                 void* __restrict__ Cv, long long sCz, int ldc,
                 const float* __restrict__ bias,
                 int M, int N, int K, float alpha, int mode) {
  __shared__ __align__(16) u16 ldsB[2][128 * 32];   // 2 x 8KB
  const int tid  = threadIdx.x;
  const int wave = tid >> 5;
  const int lane = tid & 31;
  const int z    = blockIdx.z;
  const int m0   = blockIdx.y * 128 + wave * 16;   // wave owns 16 rows
  const int n0   = blockIdx.x * 128;

  const u16* Ab = A  + (long long)z * sAz;
  const u16* Bb = Bt + (long long)z * sBz;

  v8f acc[8] = {};

  const int arow = m0 + (lane & 15);
  const int kh   = (lane >> 4) * 8;   // K-half select per WMMA 16-bit A layout

  // async-stage Bt[n0..n0+128, k0..k0+32) into LDS buffer `buf` (2x16B/thread).
  // The LDS address operand is derived from &ldsB so the object escapes and
  // the asm's memory clobber is understood to write it (low 32 bits of the
  // generic pointer are the LDS byte address per the aperture rules).
  auto stage = [&](int k0, int buf) {
#pragma unroll
    for (int i = 0; i < 2; ++i) {
      int u = tid + i * 256;                  // 0..511 16B units
      int r = u >> 2;
      int c = (u & 3) * 8;
      const u16* src = &Bb[(long long)(n0 + r) * ldb + k0 + c];
      u16* dstp = &ldsB[buf][r * 32 + c];
      unsigned dst = (unsigned)(unsigned long long)dstp;   // LDS byte address
      asm volatile("global_load_async_to_lds_b128 %0, %1, off"
                   :: "v"(dst), "v"(src) : "memory");
    }
  };

  stage(0, 0);
  asm volatile("s_wait_asynccnt 0x0" ::: "memory");
  __syncthreads();

  int cur = 0;
  for (int k0 = 0; k0 < K; k0 += 32, cur ^= 1) {
    if (k0 + 32 < K) {
      stage(k0 + 32, cur ^ 1);                        // DMA next tile (async)
      __builtin_prefetch(&Ab[(long long)arow * lda + k0 + 32], 0, 3);
    }

    const long long abase = (long long)arow * lda + k0;
    v16bf afrag = make_frag(*(const v8bf*)&Ab[abase + kh],
                            *(const v8bf*)&Ab[abase + 16 + kh]);

    // preload all 8 B fragments so ds_loads issue as a clause,
    // then run the 8 WMMAs back-to-back
    v16bf bfrag[8];
#pragma unroll
    for (int nt = 0; nt < 8; ++nt) {
      const u16* bp = &ldsB[cur][(nt * 16 + (lane & 15)) * 32];
      bfrag[nt] = make_frag(*(const v8bf*)&bp[kh],
                            *(const v8bf*)&bp[16 + kh]);
    }
#pragma unroll
    for (int nt = 0; nt < 8; ++nt)
      acc[nt] = __builtin_amdgcn_wmma_f32_16x16x32_bf16(
          false, afrag, false, bfrag[nt], (short)0, acc[nt], false, false);

    asm volatile("s_wait_asynccnt 0x0" ::: "memory");  // next tile landed
    __syncthreads();                                   // all waves done w/ cur
  }

  // C/D layout: VGPR r -> (M = r + 8*(lane>>4), N = lane&15)
  const int rowoff = (lane >> 4) * 8;
  const int coll   = lane & 15;
  if (mode == 2) {
    u16* Cb = (u16*)Cv + (long long)z * sCz;
#pragma unroll
    for (int nt = 0; nt < 8; ++nt) {
      int col = n0 + nt * 16 + coll;
#pragma unroll
      for (int r = 0; r < 8; ++r) {
        int m = m0 + rowoff + r;
        float s = acc[nt][r] * alpha;
        s = (s > 0.0f) ? s * s : 0.0f;     // relu^2
        if (col > m) s = 0.0f;             // causal (within-group) mask
        Cb[(long long)m * ldc + col] = f2bf(s);
      }
    }
  } else {
    float* Cf = (float*)Cv + (long long)z * sCz;
#pragma unroll
    for (int nt = 0; nt < 8; ++nt) {
      int col = n0 + nt * 16 + coll;
      float bv = (mode == 1) ? bias[col] : 0.0f;
#pragma unroll
      for (int r = 0; r < 8; ++r) {
        int m = m0 + rowoff + r;
        long long idx = (long long)m * ldc + col;
        float val = acc[nt][r];
        if (mode == 0)      Cf[idx] = val * alpha;
        else if (mode == 1) { float t = val + bv; Cf[idx] = t / (1.0f + __expf(-t)); }
        else                Cf[idx] += val;       // mode 3
      }
    }
  }
}

// ---------------------------------------------------------------- prep: token shift + LN (both branches)
__global__ void __launch_bounds__(256)
prep_shift_ln_kernel(const float* __restrict__ x,
                     const float* __restrict__ gh, const float* __restrict__ bh,
                     const float* __restrict__ gq, const float* __restrict__ bq,
                     u16* __restrict__ Ah, u16* __restrict__ Aq) {
  const int t = blockIdx.x;            // token 0..16383
  const int b = t >> 11, n = t & 2047;
  const int tid = threadIdx.x;
  __shared__ float vals[512];
  __shared__ float red[256];

  float local = 0.0f;
  for (int c = tid; c < 512; c += 256) {
    float v;
    if (c < 256) v = (n > 0) ? x[((long long)b * 2048 + n - 1) * 512 + c] : 0.0f;
    else         v = x[((long long)b * 2048 + n) * 512 + c];
    vals[c] = v; local += v;
  }
  red[tid] = local; __syncthreads();
  for (int s = 128; s > 0; s >>= 1) { if (tid < s) red[tid] += red[tid + s]; __syncthreads(); }
  float mu = red[0] * (1.0f / 512.0f);
  __syncthreads();
  float lv = 0.0f;
  for (int c = tid; c < 512; c += 256) { float d = vals[c] - mu; lv += d * d; }
  red[tid] = lv; __syncthreads();
  for (int s = 128; s > 0; s >>= 1) { if (tid < s) red[tid] += red[tid + s]; __syncthreads(); }
  float rs = rsqrtf(red[0] * (1.0f / 512.0f) + 1e-5f);
  for (int c = tid; c < 512; c += 256) {
    float nv = (vals[c] - mu) * rs;
    Ah[(long long)t * 512 + c] = f2bf(nv * gh[c] + bh[c]);
    Aq[(long long)t * 512 + c] = f2bf(nv * gq[c] + bq[c]);
  }
}

// ---------------------------------------------------------------- gate + LN(1024) -> bf16
__global__ void __launch_bounds__(256)
gate_ln_kernel(const float* __restrict__ attv, const float* __restrict__ attu,
               const float* __restrict__ hid,
               const float* __restrict__ go, const float* __restrict__ bo,
               u16* __restrict__ Ao) {
  const int t = blockIdx.x;
  const int tid = threadIdx.x;
  __shared__ float vals[1024];
  __shared__ float red[256];
  float local = 0.0f;
  for (int e = tid; e < 1024; e += 256) {
    float v  = hid[(long long)t * 2048 + e];
    float u  = hid[(long long)t * 2048 + 1024 + e];
    float av = attv[(long long)t * 1024 + e];
    float au = attu[(long long)t * 1024 + e];
    float g  = au * v * (1.0f / (1.0f + __expf(-av * u)));
    vals[e] = g; local += g;
  }
  red[tid] = local; __syncthreads();
  for (int s = 128; s > 0; s >>= 1) { if (tid < s) red[tid] += red[tid + s]; __syncthreads(); }
  float mu = red[0] * (1.0f / 1024.0f);
  __syncthreads();
  float lv = 0.0f;
  for (int e = tid; e < 1024; e += 256) { float d = vals[e] - mu; lv += d * d; }
  red[tid] = lv; __syncthreads();
  for (int s = 128; s > 0; s >>= 1) { if (tid < s) red[tid] += red[tid + s]; __syncthreads(); }
  float rs = rsqrtf(red[0] * (1.0f / 1024.0f) + 1e-5f);
  for (int e = tid; e < 1024; e += 256) {
    float nv = (vals[e] - mu) * rs;
    Ao[(long long)t * 1024 + e] = f2bf(nv * go[e] + bo[e]);
  }
}

// ---------------------------------------------------------------- depthwise conv K=17, SAME, + residuals
__global__ void __launch_bounds__(256)
dwconv_kernel(const float* __restrict__ h, const float* __restrict__ dw,
              const float* __restrict__ resid, float* __restrict__ out,
              int Ntime, int C, long long total) {
  long long i = (long long)blockIdx.x * 256 + threadIdx.x;
  if (i >= total) return;
  int c = (int)(i % C);
  long long tc = i / C;
  int tt = (int)(tc % Ntime);
  long long bidx = tc / Ntime;
  float acc = 0.0f;
#pragma unroll
  for (int k = 0; k < 17; ++k) {
    int ts = tt + k - 8;
    if (ts >= 0 && ts < Ntime)
      acc += h[(bidx * Ntime + ts) * C + c] * dw[k * C + c];
  }
  float r = (resid != nullptr) ? resid[i] : 0.0f;
  out[i] = h[i] + acc + r;
}

// ---------------------------------------------------------------- W fp32 [K,N] -> bf16 Wt [N,K]
__global__ void __launch_bounds__(256)
convert_transpose_kernel(const float* __restrict__ W, u16* __restrict__ Wt,
                         int Kd, int Nd) {
  long long i = (long long)blockIdx.x * 256 + threadIdx.x;
  if (i >= (long long)Kd * Nd) return;
  int k = (int)(i / Nd), n = (int)(i % Nd);
  Wt[(long long)n * Kd + k] = f2bf(W[i]);
}

// ---------------------------------------------------------------- qk -> 4 scaled heads (bf16), lin_k transposed per group
__global__ void __launch_bounds__(256)
qk4_kernel(const float* __restrict__ qk,
           const float* __restrict__ gamma, const float* __restrict__ beta,
           u16* __restrict__ qq, u16* __restrict__ lq, u16* __restrict__ kk,
           u16* __restrict__ lkT) {
  long long i = (long long)blockIdx.x * 256 + threadIdx.x;  // 16384*128
  if (i >= 16384ll * 128) return;
  int d = (int)(i & 127);
  long long t = i >> 7;
  float v = qk[i];
  qq[i] = f2bf(v * gamma[0 * 128 + d] + beta[0 * 128 + d]);  // quad_q
  lq[i] = f2bf(v * gamma[1 * 128 + d] + beta[1 * 128 + d]);  // lin_q
  kk[i] = f2bf(v * gamma[2 * 128 + d] + beta[2 * 128 + d]);  // quad_k
  int z = (int)(t >> 8), j = (int)(t & 255);
  lkT[((long long)z * 128 + d) * 256 + j] =
      f2bf(v * gamma[3 * 128 + d] + beta[3 * 128 + d]);      // lin_k^T [grp][d][j]
}

// ---------------------------------------------------------------- hid -> vT/uT bf16 [grp][e][j]
__global__ void __launch_bounds__(256)
vu_transpose_kernel(const float* __restrict__ hid,
                    u16* __restrict__ vT, u16* __restrict__ uT) {
  long long i = (long long)blockIdx.x * 256 + threadIdx.x;  // 16384*1024
  if (i >= 16384ll * 1024) return;
  int e = (int)(i & 1023);
  long long t = i >> 10;
  int z = (int)(t >> 8), j = (int)(t & 255);
  long long o = ((long long)z * 1024 + e) * 256 + j;
  vT[o] = f2bf(hid[t * 2048 + e]);
  uT[o] = f2bf(hid[t * 2048 + 1024 + e]);
}

// ---------------------------------------------------------------- exclusive cumsum over groups + transpose to bf16 [bg][which][e][d]
__global__ void __launch_bounds__(256)
kv_cumsum_kernel(const float* __restrict__ kv_v, const float* __restrict__ kv_u,
                 u16* __restrict__ kvcumT) {
  long long i = (long long)blockIdx.x * 256 + threadIdx.x;  // 8*128*1024
  if (i >= 8ll * 128 * 1024) return;
  int b = (int)(i >> 17);
  int d = (int)((i >> 10) & 127);
  int e = (int)(i & 1023);
  float accv = 0.0f, accu = 0.0f;
  for (int g = 0; g < 8; ++g) {
    long long zz = (long long)b * 8 + g;
    kvcumT[((zz * 2 + 0) * 1024 + e) * 128 + d] = f2bf(accv);
    kvcumT[((zz * 2 + 1) * 1024 + e) * 128 + d] = f2bf(accu);
    accv += kv_v[(zz * 128 + d) * 1024 + e];
    accu += kv_u[(zz * 128 + d) * 1024 + e];
  }
}

// ---------------------------------------------------------------- host
extern "C" void kernel_launch(void* const* d_in, const int* in_sizes, int n_in,
                              void* d_out, int out_size, void* d_ws, size_t ws_size,
                              hipStream_t stream) {
  (void)in_sizes; (void)n_in; (void)out_size; (void)ws_size;
  const float* x      = (const float*)d_in[0];
  const float* ln_h_g = (const float*)d_in[1];
  const float* ln_h_b = (const float*)d_in[2];
  const float* W_h    = (const float*)d_in[3];
  const float* b_h    = (const float*)d_in[4];
  const float* dw_h   = (const float*)d_in[5];
  const float* ln_q_g = (const float*)d_in[6];
  const float* ln_q_b = (const float*)d_in[7];
  const float* W_qk   = (const float*)d_in[8];
  const float* b_qk   = (const float*)d_in[9];
  const float* dw_qk  = (const float*)d_in[10];
  const float* gamma  = (const float*)d_in[11];
  const float* beta   = (const float*)d_in[12];
  const float* ln_o_g = (const float*)d_in[13];
  const float* ln_o_b = (const float*)d_in[14];
  const float* W_o    = (const float*)d_in[15];
  const float* b_o    = (const float*)d_in[16];
  const float* dw_o   = (const float*)d_in[17];

  const int TOK = 8 * 2048;  // tokens
  char* ws = (char*)d_ws;
  size_t off = 0;
  auto alloc = [&](size_t bytes) -> void* {
    void* p = ws + off;
    off += (bytes + 255) & ~(size_t)255;
    return p;
  };

  u16*   Ah      = (u16*)  alloc((size_t)TOK * 512 * 2);
  u16*   Aq      = (u16*)  alloc((size_t)TOK * 512 * 2);
  u16*   WtH     = (u16*)  alloc((size_t)2048 * 512 * 2);
  u16*   WtQ     = (u16*)  alloc((size_t)128 * 512 * 2);
  u16*   WtO     = (u16*)  alloc((size_t)512 * 1024 * 2);
  float* hid_pre = (float*)alloc((size_t)TOK * 2048 * 4);   // reused later as att_v/att_u
  float* hid     = (float*)alloc((size_t)TOK * 2048 * 4);
  float* qk_pre  = (float*)alloc((size_t)TOK * 128 * 4);
  float* qk      = (float*)alloc((size_t)TOK * 128 * 4);
  u16*   qq      = (u16*)  alloc((size_t)TOK * 128 * 2);
  u16*   lq      = (u16*)  alloc((size_t)TOK * 128 * 2);
  u16*   kk      = (u16*)  alloc((size_t)TOK * 128 * 2);
  u16*   lkT     = (u16*)  alloc((size_t)64 * 128 * 256 * 2);
  u16*   vT      = (u16*)  alloc((size_t)64 * 1024 * 256 * 2);
  u16*   uT      = (u16*)  alloc((size_t)64 * 1024 * 256 * 2);
  u16*   attn    = (u16*)  alloc((size_t)64 * 256 * 256 * 2);
  float* kv_v    = (float*)alloc((size_t)64 * 128 * 1024 * 4);
  float* kv_u    = (float*)alloc((size_t)64 * 128 * 1024 * 4);
  u16*   kvcumT  = (u16*)  alloc((size_t)64 * 2 * 1024 * 128 * 2);
  u16*   Ao      = (u16*)  alloc((size_t)TOK * 1024 * 2);
  float* h_o     = (float*)alloc((size_t)TOK * 512 * 4);
  float* att_v   = hid_pre;                       // hid_pre dead after dwconv(hid)
  float* att_u   = hid_pre + (size_t)TOK * 1024;

  auto gemm = [&](const u16* A, long long sAz, int lda,
                  const u16* Bt, long long sBz, int ldb,
                  void* C, long long sCz, int ldc,
                  const float* bias, int M, int Nc, int Kc, int Z,
                  float alpha, int mode) {
    dim3 grid(Nc / 128, M / 128, Z);
    wmma_gemm_kernel<<<grid, dim3(256), 0, stream>>>(
        A, sAz, lda, Bt, sBz, ldb, C, sCz, ldc, bias, M, Nc, Kc, alpha, mode);
  };
  auto ewgrid = [](long long total) { return dim3((unsigned)((total + 255) / 256)); };

  // 1) token shift + LN (both branch scalings)
  prep_shift_ln_kernel<<<dim3(TOK), dim3(256), 0, stream>>>(
      x, ln_h_g, ln_h_b, ln_q_g, ln_q_b, Ah, Aq);

  // 2) weight transposes to bf16 [N,K]
  convert_transpose_kernel<<<ewgrid(512ll * 2048), dim3(256), 0, stream>>>(W_h, WtH, 512, 2048);
  convert_transpose_kernel<<<ewgrid(512ll * 128),  dim3(256), 0, stream>>>(W_qk, WtQ, 512, 128);
  convert_transpose_kernel<<<ewgrid(1024ll * 512), dim3(256), 0, stream>>>(W_o, WtO, 1024, 512);

  // 3) hid_pre = silu(Ah @ W_h + b_h); then depthwise residual -> hid
  gemm(Ah, 0, 512, WtH, 0, 512, hid_pre, 0, 2048, b_h, TOK, 2048, 512, 1, 1.0f, 1);
  dwconv_kernel<<<ewgrid((long long)TOK * 2048), dim3(256), 0, stream>>>(
      hid_pre, dw_h, nullptr, hid, 2048, 2048, (long long)TOK * 2048);

  // 4) qk branch
  gemm(Aq, 0, 512, WtQ, 0, 512, qk_pre, 0, 128, b_qk, TOK, 128, 512, 1, 1.0f, 1);
  dwconv_kernel<<<ewgrid((long long)TOK * 128), dim3(256), 0, stream>>>(
      qk_pre, dw_qk, nullptr, qk, 2048, 128, (long long)TOK * 128);

  // 5) offset-scale heads + transposed operand prep
  qk4_kernel<<<ewgrid(16384ll * 128), dim3(256), 0, stream>>>(qk, gamma, beta, qq, lq, kk, lkT);
  vu_transpose_kernel<<<ewgrid(16384ll * 1024), dim3(256), 0, stream>>>(hid, vT, uT);

  // 6) quadratic attention: attn = causal(relu(QK^T/g)^2) as bf16
  gemm(qq, 256ll * 128, 128, kk, 256ll * 128, 128, attn, 256ll * 256, 256,
       nullptr, 256, 256, 128, 64, 1.0f / 256.0f, 2);
  //    quad_v/quad_u -> att buffers (write)
  gemm(attn, 256ll * 256, 256, vT, 1024ll * 256, 256, att_v, 256ll * 1024, 1024,
       nullptr, 256, 1024, 256, 64, 1.0f, 0);
  gemm(attn, 256ll * 256, 256, uT, 1024ll * 256, 256, att_u, 256ll * 1024, 1024,
       nullptr, 256, 1024, 256, 64, 1.0f, 0);

  // 7) linear attention: kv = lin_k^T @ {v,u} / g, exclusive cumsum over groups, then lin_q @ kv (+=)
  gemm(lkT, 128ll * 256, 256, vT, 1024ll * 256, 256, kv_v, 128ll * 1024, 1024,
       nullptr, 128, 1024, 256, 64, 1.0f / 256.0f, 0);
  gemm(lkT, 128ll * 256, 256, uT, 1024ll * 256, 256, kv_u, 128ll * 1024, 1024,
       nullptr, 128, 1024, 256, 64, 1.0f / 256.0f, 0);
  kv_cumsum_kernel<<<ewgrid(8ll * 128 * 1024), dim3(256), 0, stream>>>(kv_v, kv_u, kvcumT);
  gemm(lq, 256ll * 128, 128, kvcumT,              2ll * 1024 * 128, 128,
       att_v, 256ll * 1024, 1024, nullptr, 256, 1024, 128, 64, 1.0f, 3);
  gemm(lq, 256ll * 128, 128, kvcumT + 1024ll * 128, 2ll * 1024 * 128, 128,
       att_u, 256ll * 1024, 1024, nullptr, 256, 1024, 128, 64, 1.0f, 3);

  // 8) gate + LN(1024) -> bf16 A_o
  gate_ln_kernel<<<dim3(TOK), dim3(256), 0, stream>>>(att_v, att_u, hid, ln_o_g, ln_o_b, Ao);

  // 9) to_out GEMM + SiLU, then depthwise conv residual + x residual -> d_out
  gemm(Ao, 0, 1024, WtO, 0, 1024, h_o, 0, 512, b_o, TOK, 512, 1024, 1, 1.0f, 1);
  dwconv_kernel<<<ewgrid((long long)TOK * 512), dim3(256), 0, stream>>>(
      h_o, dw_o, x, (float*)d_out, 2048, 512, (long long)TOK * 512);
}
